// ModelTorch_65335042507138
// MI455X (gfx1250) — matrix operand order repaired and checked by reading, not compiled, not run on hardware
//
#include <hip/hip_runtime.h>
#include <hip/hip_bf16.h>
#include <stdint.h>

typedef __attribute__((ext_vector_type(16))) _Float16 v16h;
typedef __attribute__((ext_vector_type(8)))  float    v8f;

__device__ __forceinline__ float readlane_f32(float x, int lane) {
  return __uint_as_float(__builtin_amdgcn_readlane(__float_as_uint(x), lane));
}

// ---------------------------------------------------------------------------
// Kernel 1: per-block (256 sequences) inclusive scan of accept_lens using WMMA.
//   One wave (32 threads) per block. Values staged to LDS with the CDNA5 async
//   global->LDS path, then scanned with two v_wmma_f32_16x16x32_f16 ops:
//     P = V * U      (U upper-triangular ones -> row-wise prefix of 16x16 tile)
//     S = W * T + P  (W strictly-lower-triangular ones, T = row totals)
//   Exact: accept < 128 (f16-exact), row totals <= 2032 (f16-exact), block
//   totals <= 32512 (f32-exact).
// ---------------------------------------------------------------------------
__global__ void __launch_bounds__(32)
scan256_wmma(const int* __restrict__ acc, int* __restrict__ local_incl,
             int* __restrict__ bsum) {
  __shared__ int sacc[256];
  const int lane = threadIdx.x;          // 0..31, wave32
  const int base = blockIdx.x * 256;

  // ---- async stage 256 ints (1 KB) global -> LDS, tracked by ASYNCcnt ----
  unsigned ldsAddr = (unsigned)(uintptr_t)sacc + (unsigned)lane * 16u;
  unsigned vOff    = (unsigned)lane * 16u;
  const int* gptr  = acc + base;         // uniform -> SGPR pair
  asm volatile(
      "global_load_async_to_lds_b128 %0, %1, %2\n\t"
      "global_load_async_to_lds_b128 %0, %1, %2 offset:512\n\t"
      "s_wait_asynccnt 0"
      :
      : "v"(ldsAddr), "v"(vOff), "s"(gptr)
      : "memory");

  // ---- build A = V (16x16 values, K-cols 16..31 zero) ----
  // A layout: lanes 0-15 hold row M=lane, K=0..7 in elts 0..7;
  //           lanes 16-31 hold row M=lane-16, K=8..15 in elts 0..7.
  const int rowStart = (lane < 16) ? (lane * 16) : ((lane - 16) * 16 + 8);
  v16h A, Bm;
#pragma unroll
  for (int t = 0; t < 8; ++t) A[t] = (_Float16)(float)sacc[rowStart + t];
#pragma unroll
  for (int t = 8; t < 16; ++t) A[t] = (_Float16)0.0f;

  // ---- B = U: U[k][n] = (k <= n), rows K>=16 zero (elts 8..15) ----
  // B layout: lane holds col N=lane%16, K=8*(lane/16)+elt for elts 0..7 (elts 8..15 -> K>=16, zero).
#pragma unroll
  for (int t = 0; t < 16; ++t)
    Bm[t] = (_Float16)((t < 8 && (8 * (lane >> 4) + t) <= (lane & 15)) ? 1.0f : 0.0f);

  v8f Cz = {};
  v8f P = __builtin_amdgcn_wmma_f32_16x16x32_f16(false, A, false, Bm,
                                                 (short)0, Cz, false, false);

  // ---- row totals T[r] = P[r][15] via v_readlane (no LDS traffic) ----
  // D layout: VGPR v holds M=v (lanes 0-15, N=lane) / M=v+8 (lanes 16-31).
  float T[16];
#pragma unroll
  for (int r = 0; r < 8; ++r) {
    T[r]     = readlane_f32(P[r], 15);
    T[r + 8] = readlane_f32(P[r], 31);
  }

  // ---- A2 = W: W[m][k] = (k < m);  B2[k][n] = T[k] broadcast ----
  v16h A2, B2;
  const int M     = (lane < 16) ? lane : (lane - 16);
  const int kbase = (lane < 16) ? 0 : 8;
#pragma unroll
  for (int t = 0; t < 8; ++t)
    A2[t] = (_Float16)(((kbase + t) < M) ? 1.0f : 0.0f);
#pragma unroll
  for (int t = 8; t < 16; ++t) A2[t] = (_Float16)0.0f;
#pragma unroll
  for (int t = 0; t < 16; ++t)
    B2[t] = (t < 8) ? (_Float16)((lane < 16) ? T[t & 7] : T[(t & 7) + 8]) : (_Float16)0.0f;

  // S[m][n] = sum_k W[m][k]*T[k] + P[m][n]  == full 256-element inclusive scan
  v8f S = __builtin_amdgcn_wmma_f32_16x16x32_f16(false, A2, false, B2,
                                                 (short)0, P, false, false);

  // ---- write back: VGPR v, lane l -> element 16*M + N ----
  const int n    = lane & 15;
  const int mofs = (lane < 16) ? 0 : 8;
#pragma unroll
  for (int v = 0; v < 8; ++v) {
    const int idx = (v + mofs) * 16 + n;
    local_incl[base + idx] = (int)S[v];
  }
  if (lane == 31) bsum[blockIdx.x] = (int)S[7];   // scan[255] = block total
}

// ---------------------------------------------------------------------------
// Kernel 2: inclusive scan of the (<=512) per-block sums, single block.
// ---------------------------------------------------------------------------
__global__ void __launch_bounds__(512)
scan_bsums(int* __restrict__ bsum, int nb) {
  __shared__ int s[512];
  const int t = threadIdx.x;
  s[t] = (t < nb) ? bsum[t] : 0;
  __syncthreads();
#pragma unroll
  for (int d = 1; d < 512; d <<= 1) {
    const int x = (t >= d) ? s[t - d] : 0;
    __syncthreads();
    s[t] += x;
    __syncthreads();
  }
  if (t < nb) bsum[t] = s[t];
}

// ---------------------------------------------------------------------------
// Kernel 3: gather new_verified_id and wave-cooperatively fill positions.
//   positions[start_b + j] = seq_b - acc_b + j   (coalesced, NT stores)
//   Per-segment parameters broadcast with v_readlane (dynamic uniform lane)
//   so loop control and store base are scalar; no LDS/ds_bpermute in the
//   hot loop that streams the ~33 MB positions output.
// ---------------------------------------------------------------------------
__global__ void __launch_bounds__(256)
finalize_fill(const float* __restrict__ vid, const int* __restrict__ seq,
              const int* __restrict__ acc, const int* __restrict__ local_incl,
              const int* __restrict__ bsum_incl, float* __restrict__ out_pos,
              float* __restrict__ out_vid, int nSeq) {
  const int tid  = threadIdx.x;
  const int b    = blockIdx.x * 256 + tid;
  const int prev = (blockIdx.x == 0) ? 0 : bsum_incl[blockIdx.x - 1];

  int a = 0, off = 0, basev = 0;
  if (b < nSeq) {
    a     = acc[b];
    off   = local_incl[b] + prev;               // inclusive global offset
    basev = seq[b] - a;                         // seq_len - accept_len
    out_vid[b] = (a > 0) ? vid[off - 1] : 0.0f;
  }
  const int start = off - a;
  const int lane  = tid & 31;

  // wave-cooperative ragged fill: all 32 lanes stripe each segment in turn
#pragma unroll 1
  for (int s = 0; s < 32; ++s) {
    const int st = __builtin_amdgcn_readlane(start, s);   // scalar broadcast
    const int cn = __builtin_amdgcn_readlane(a, s);
    const int bs = __builtin_amdgcn_readlane(basev, s);
    for (int j = lane; j < cn; j += 32)
      __builtin_nontemporal_store((float)(bs + j), &out_pos[st + j]);
  }
}

// ---------------------------------------------------------------------------
// Host launcher
//   d_in[0] = verified_id (float, T)   d_in[1] = seq_lens (int, B)
//   d_in[2] = accept_lens (int, B)
//   d_out   = [positions (T floats) | new_verified_id (B floats)]
// ---------------------------------------------------------------------------
extern "C" void kernel_launch(void* const* d_in, const int* in_sizes, int n_in,
                              void* d_out, int out_size, void* d_ws,
                              size_t ws_size, hipStream_t stream) {
  const float* vid = (const float*)d_in[0];
  const int*   seq = (const int*)d_in[1];
  const int*   acc = (const int*)d_in[2];
  const int T = in_sizes[0];
  const int B = in_sizes[1];
  const int nb = (B + 255) / 256;               // 512 for B=131072

  int* ws_local = (int*)d_ws;                   // B ints: per-elem local scan
  int* ws_bsum  = ws_local + B;                 // nb ints: block sums / prefix

  float* out_pos = (float*)d_out;
  float* out_vid = out_pos + T;

  scan256_wmma<<<nb, 32, 0, stream>>>(acc, ws_local, ws_bsum);
  scan_bsums<<<1, 512, 0, stream>>>(ws_bsum, nb);
  finalize_fill<<<nb, 256, 0, stream>>>(vid, seq, acc, ws_local, ws_bsum,
                                        out_pos, out_vid, B);
}
